// SingleHeadAttention_76098230550846
// MI455X (gfx1250) — matrix operand order repaired and checked by reading, compiled, not run on hardware
//
#include <hip/hip_runtime.h>
#include <hip/hip_bf16.h>

typedef _Float16 v16h __attribute__((ext_vector_type(16)));
typedef _Float16 v8h  __attribute__((ext_vector_type(8)));
typedef float    v8f  __attribute__((ext_vector_type(8)));
typedef float    v4f  __attribute__((ext_vector_type(4)));

#define B_    8
#define T_    2048
#define E_    1024
#define PADH  40    // f16 LDS row stride in halfs  (80 B, 16B-aligned, conflict-free)
#define PADF  36    // f32 LDS row stride in floats (144 B, 16B-aligned, conflict-free)

__device__ __forceinline__ v8f wmma_f16(v16h a, v16h b, v8f c) {
    return __builtin_amdgcn_wmma_f32_16x16x32_f16(
        false, a, false, b, (short)0, c, false, false);
}

__device__ __forceinline__ v16h cat16(v8h lo, v8h hi) {
    v16h v;
#pragma unroll
    for (int i = 0; i < 8; ++i) { v[i] = lo[i]; v[8 + i] = hi[i]; }
    return v;
}

// ---- gfx1250 async global->LDS copy (ASYNCcnt) ----
__device__ __forceinline__ void async_copy_b128(const void* gptr, void* lptr) {
    // low 32 bits of a generic pointer into LDS are the LDS byte offset
    unsigned loff = (unsigned)(unsigned long long)lptr;
    asm volatile("global_load_async_to_lds_b128 %0, %1, off"
                 :: "v"(loff), "v"((unsigned long long)gptr)
                 : "memory");
}
__device__ __forceinline__ void wait_async0() {
    asm volatile("s_wait_asynccnt 0" ::: "memory");
}
// Raw split workgroup barrier: no implicit LOADcnt/STOREcnt drain (unlike
// __syncthreads). LDS visibility is guaranteed by s_wait_asynccnt (async
// writes) and by the fact that all ds_load results are consumed by WMMAs
// before the barrier (DScnt waited by data dependence).
__device__ __forceinline__ void block_barrier() {
    asm volatile("s_barrier_signal -1\n\ts_barrier_wait -1" ::: "memory");
}

// ===================== fragment / staging helpers =====================
// Wave tile: 32(M) x 64(N), k-step 32.
// A frag (16x32 f16): lane row = lane&15, hh = lane>>4
//   elems 0..7 = A[row, k+hh*8..+7], elems 8..15 = A[row, k+16+hh*8..+7]
// B frag (32x16): lane col = lane&15, elems 0..7 = B[col, k+hh*8..+7], 8..15 = B[col, k+16+hh*8..+7]

__device__ __forceinline__ void load_a_f16(v8h (&a)[2][2],
        const _Float16* __restrict__ A, int lda, int m0, int k, int lane) {
    const int row = lane & 15;
    const int hh  = lane >> 4;
#pragma unroll
    for (int u = 0; u < 2; ++u) {
        const _Float16* p = A + (size_t)(m0 + u * 16 + row) * lda + k + hh * 8;
        a[u][0] = *(const v8h*)p;
        a[u][1] = *(const v8h*)(p + 16);
    }
}

__device__ __forceinline__ void load_a_f32(v4f (&a)[2][4],
        const float* __restrict__ A, int lda, int m0, int k, int lane) {
    const int row = lane & 15;
    const int hh  = lane >> 4;
#pragma unroll
    for (int u = 0; u < 2; ++u) {
        const float* p = A + (size_t)(m0 + u * 16 + row) * lda + k + hh * 8;
        __builtin_prefetch(p + 64, 0, 3);      // WGP-scope prefetch, 2 k-steps ahead
        a[u][0] = *(const v4f*)(p + 0);
        a[u][1] = *(const v4f*)(p + 4);
        a[u][2] = *(const v4f*)(p + 16);
        a[u][3] = *(const v4f*)(p + 20);
    }
}

// stage 64 N-rows x 32 k-halfs of B (f16) into LDS: 1 async b128 per thread
__device__ __forceinline__ void stage_b_f16(_Float16* dst,
        const _Float16* __restrict__ Bm, int ldb, int n0, int k, int tid) {
    const int row = tid >> 2;     // 0..63
    const int ch  = tid & 3;      // 4 x 8 halfs = 32 halfs per row
    const _Float16* g = Bm + (size_t)(n0 + row) * ldb + k + ch * 8;
    async_copy_b128(g, dst + row * PADH + ch * 8);
}

// stage 64 rows x 32 k-floats of B (f32) into LDS: 2 async b128 per thread
__device__ __forceinline__ void stage_b_f32(float* dst,
        const float* __restrict__ Bm, int ldb, int n0, int k, int tid) {
    const int row = tid >> 2;
    const int ch  = tid & 3;      // 4 x 8 floats = 32 floats per row
    const float* g = Bm + (size_t)(n0 + row) * ldb + k + ch * 8;
    float* l = dst + row * PADF + ch * 8;
    async_copy_b128(g, l);
    async_copy_b128(g + 4, l + 4);
}

__device__ __forceinline__ void compute_lds_f16(const _Float16* Bl,
        const v8h (&a)[2][2], v8f (&acc)[2][4], int lane) {
    const int col = lane & 15;
    const int hh  = lane >> 4;
    v16h bf[4];
#pragma unroll
    for (int t = 0; t < 4; ++t) {
        const _Float16* p = Bl + (t * 16 + col) * PADH + hh * 8;
        bf[t] = cat16(*(const v8h*)p, *(const v8h*)(p + 16));
    }
#pragma unroll
    for (int u = 0; u < 2; ++u) {
        v16h av = cat16(a[u][0], a[u][1]);
#pragma unroll
        for (int t = 0; t < 4; ++t) acc[u][t] = wmma_f16(av, bf[t], acc[u][t]);
    }
}

__device__ __forceinline__ void compute_lds_f32(const float* Bl,
        const v4f (&a)[2][4], v8f (&acc)[2][4], int lane) {
    const int col = lane & 15;
    const int hh  = lane >> 4;
    v16h bf[4];
#pragma unroll
    for (int t = 0; t < 4; ++t) {
        const float* p = Bl + (t * 16 + col) * PADF + hh * 8;
        v16h b;
#pragma unroll
        for (int j = 0; j < 4; ++j) {
            v4f x = *(const v4f*)(p + 4 * j + (j >> 1) * 8);
#pragma unroll
            for (int i = 0; i < 4; ++i) b[j * 4 + i] = (_Float16)x[i];
        }
        bf[t] = b;
    }
#pragma unroll
    for (int u = 0; u < 2; ++u) {
        v16h av;
#pragma unroll
        for (int j = 0; j < 4; ++j)
#pragma unroll
            for (int i = 0; i < 4; ++i) av[j * 4 + i] = (_Float16)a[u][j][i];
#pragma unroll
        for (int t = 0; t < 4; ++t) acc[u][t] = wmma_f16(av, bf[t], acc[u][t]);
    }
}

// ============ Kernel 1: projection  C = X * W^T  (f32 in, f16 out) ============
__global__ __launch_bounds__(256)
void attn_proj_kernel(const float* __restrict__ X, const float* __restrict__ W,
                      _Float16* __restrict__ Out, float scale, int transposeV) {
    __shared__ float shB[2][64 * PADF];
    const int tid  = threadIdx.x;
    const int lane = tid & 31;
    const int wave = tid >> 5;
    const int n0 = blockIdx.x * 64;
    const int m0 = blockIdx.y * 256 + wave * 32;

    v8f z = {};
    v8f acc[2][4];
#pragma unroll
    for (int u = 0; u < 2; ++u)
#pragma unroll
        for (int t = 0; t < 4; ++t) acc[u][t] = z;

    v4f a0[2][4], a1[2][4];
    stage_b_f32(shB[0], W, E_, n0, 0, tid);
    load_a_f32(a0, X, E_, m0, 0, lane);
    wait_async0(); block_barrier();

    int k = 0;
    for (;;) {
        int kn = k + 32;
        if (kn < E_) { stage_b_f32(shB[1], W, E_, n0, kn, tid);
                       load_a_f32(a1, X, E_, m0, kn, lane); }
        compute_lds_f32(shB[0], a0, acc, lane);
        if (kn >= E_) break;
        wait_async0(); block_barrier();
        k = kn; kn = k + 32;
        if (kn < E_) { stage_b_f32(shB[0], W, E_, n0, kn, tid);
                       load_a_f32(a0, X, E_, m0, kn, lane); }
        compute_lds_f32(shB[1], a1, acc, lane);
        if (kn >= E_) break;
        wait_async0(); block_barrier();
        k = kn;
    }

    const int col   = lane & 15;
    const int rbase = (lane >> 4) * 8;
#pragma unroll
    for (int u = 0; u < 2; ++u)
#pragma unroll
    for (int t = 0; t < 4; ++t)
#pragma unroll
    for (int r = 0; r < 8; ++r) {
        const int m = m0 + u * 16 + r + rbase;
        const int n = n0 + t * 16 + col;
        const float val = acc[u][t][r] * scale;
        if (transposeV) {
            const int bi = m >> 11;
            const int il = m & (T_ - 1);
            Out[((size_t)bi * E_ + n) * T_ + il] = (_Float16)val;
        } else {
            Out[(size_t)m * E_ + n] = (_Float16)val;
        }
    }
}

// ============ Kernel 2: S = Qh * Kh^T (block-causal) ============
__global__ __launch_bounds__(256)
void attn_scores_kernel(const _Float16* __restrict__ Qh,
                        const _Float16* __restrict__ Kh,
                        _Float16* __restrict__ S) {
    const int bi  = blockIdx.z;
    const int j0b = blockIdx.x * 64;
    const int i0b = blockIdx.y * 256;
    if (j0b > i0b + 255) return;              // block-uniform causal skip

    __shared__ _Float16 shB[2][64 * PADH];
    const int tid  = threadIdx.x;
    const int lane = tid & 31;
    const int wave = tid >> 5;
    const int i0 = i0b + wave * 32;

    const _Float16* Qb = Qh + (size_t)bi * T_ * E_;
    const _Float16* Kb = Kh + (size_t)bi * T_ * E_;
    _Float16* Sb = S + (size_t)bi * T_ * T_;

    v8f z = {};
    v8f acc[2][4];
#pragma unroll
    for (int u = 0; u < 2; ++u)
#pragma unroll
        for (int t = 0; t < 4; ++t) acc[u][t] = z;

    v8h a0[2][2], a1[2][2];
    stage_b_f16(shB[0], Kb, E_, j0b, 0, tid);
    load_a_f16(a0, Qb, E_, i0, 0, lane);
    wait_async0(); block_barrier();

    int k = 0;
    for (;;) {
        int kn = k + 32;
        if (kn < E_) { stage_b_f16(shB[1], Kb, E_, j0b, kn, tid);
                       load_a_f16(a1, Qb, E_, i0, kn, lane); }
        compute_lds_f16(shB[0], a0, acc, lane);
        if (kn >= E_) break;
        wait_async0(); block_barrier();
        k = kn; kn = k + 32;
        if (kn < E_) { stage_b_f16(shB[0], Kb, E_, j0b, kn, tid);
                       load_a_f16(a0, Qb, E_, i0, kn, lane); }
        compute_lds_f16(shB[1], a1, acc, lane);
        if (kn >= E_) break;
        wait_async0(); block_barrier();
        k = kn;
    }

    const int col   = lane & 15;
    const int rbase = (lane >> 4) * 8;
#pragma unroll
    for (int u = 0; u < 2; ++u)
#pragma unroll
    for (int t = 0; t < 4; ++t)
#pragma unroll
    for (int r = 0; r < 8; ++r) {
        const int i = i0 + u * 16 + r + rbase;
        const int j = j0b + t * 16 + col;
        Sb[(size_t)i * T_ + j] = (_Float16)acc[u][t][r];
    }
}

// ============ Kernel 3: row softmax with causal mask, in place ============
__global__ __launch_bounds__(256)
void attn_softmax_kernel(_Float16* __restrict__ S) {
    const int lane = threadIdx.x & 31;
    const int wave = threadIdx.x >> 5;
    const int row  = blockIdx.x * 8 + wave;
    const int i    = row & (T_ - 1);
    _Float16* Srow = S + (size_t)row * T_;

    float mx = -INFINITY;
    for (int j = lane; j <= i; j += 32) mx = fmaxf(mx, (float)Srow[j]);
#pragma unroll
    for (int off = 16; off > 0; off >>= 1) mx = fmaxf(mx, __shfl_xor(mx, off, 32));

    float sum = 0.0f;
    for (int j = lane; j <= i; j += 32) sum += __expf((float)Srow[j] - mx);
#pragma unroll
    for (int off = 16; off > 0; off >>= 1) sum += __shfl_xor(sum, off, 32);

    const float inv = 1.0f / sum;
    for (int j = lane; j < T_; j += 32) {
        float p = (j <= i) ? __expf((float)Srow[j] - mx) * inv : 0.0f;
        Srow[j] = (_Float16)p;
    }
}

// ============ Kernel 4: O = P * V  (B from transposed Vt) ============
__global__ __launch_bounds__(256)
void attn_out_kernel(const _Float16* __restrict__ P,
                     const _Float16* __restrict__ Vt,
                     float* __restrict__ O) {
    __shared__ _Float16 shB[2][64 * PADH];
    const int tid  = threadIdx.x;
    const int lane = tid & 31;
    const int wave = tid >> 5;
    const int bi = blockIdx.z;
    const int n0 = blockIdx.x * 64;
    const int i0b = blockIdx.y * 256;
    const int i0  = i0b + wave * 32;

    const _Float16* Pb = P  + (size_t)bi * T_ * T_;
    const _Float16* Vb = Vt + (size_t)bi * E_ * T_;

    const int kend = i0b + 256;                // block-uniform causal extent (<= T_)

    v8f z = {};
    v8f acc[2][4];
#pragma unroll
    for (int u = 0; u < 2; ++u)
#pragma unroll
        for (int t = 0; t < 4; ++t) acc[u][t] = z;

    v8h a0[2][2], a1[2][2];
    stage_b_f16(shB[0], Vb, T_, n0, 0, tid);
    load_a_f16(a0, Pb, T_, i0, 0, lane);
    wait_async0(); block_barrier();

    int k = 0;
    for (;;) {
        int kn = k + 32;
        if (kn < kend) { stage_b_f16(shB[1], Vb, T_, n0, kn, tid);
                         load_a_f16(a1, Pb, T_, i0, kn, lane); }
        compute_lds_f16(shB[0], a0, acc, lane);
        if (kn >= kend) break;
        wait_async0(); block_barrier();
        k = kn; kn = k + 32;
        if (kn < kend) { stage_b_f16(shB[0], Vb, T_, n0, kn, tid);
                         load_a_f16(a0, Pb, T_, i0, kn, lane); }
        compute_lds_f16(shB[1], a1, acc, lane);
        if (kn >= kend) break;
        wait_async0(); block_barrier();
        k = kn;
    }

    const int col   = lane & 15;
    const int rbase = (lane >> 4) * 8;
#pragma unroll
    for (int u = 0; u < 2; ++u)
#pragma unroll
    for (int t = 0; t < 4; ++t)
#pragma unroll
    for (int r = 0; r < 8; ++r) {
        const int i = i0 + u * 16 + r + rbase;
        const int n = n0 + t * 16 + col;
        O[((size_t)bi * T_ + i) * E_ + n] = acc[u][t][r];
    }
}

extern "C" void kernel_launch(void* const* d_in, const int* in_sizes, int n_in,
                              void* d_out, int out_size, void* d_ws, size_t ws_size,
                              hipStream_t stream) {
    const float* q  = (const float*)d_in[0];
    const float* k  = (const float*)d_in[1];
    const float* v  = (const float*)d_in[2];
    const float* Wq = (const float*)d_in[3];
    const float* Wk = (const float*)d_in[4];
    const float* Wv = (const float*)d_in[5];
    float* out = (float*)d_out;

    const size_t nQKV = (size_t)B_ * T_ * E_;
    _Float16* Qh = (_Float16*)d_ws;
    _Float16* Kh = Qh + nQKV;
    _Float16* Vt = Kh + nQKV;                  // [b][E][T]
    _Float16* S  = Vt + nQKV;                  // [b][T][T] -> becomes P

    const float scale = 0.17677669529663689f;  // 1 / 1024^(1/4)

    dim3 blk(256);
    dim3 gProj(E_ / 64, (B_ * T_) / 256);
    attn_proj_kernel<<<gProj, blk, 0, stream>>>(q, Wq, Qh, scale, 0);
    attn_proj_kernel<<<gProj, blk, 0, stream>>>(k, Wk, Kh, scale, 0);
    attn_proj_kernel<<<gProj, blk, 0, stream>>>(v, Wv, Vt, 1.0f, 1);

    dim3 gS(T_ / 64, T_ / 256, B_);
    attn_scores_kernel<<<gS, blk, 0, stream>>>(Qh, Kh, S);

    attn_softmax_kernel<<<dim3((B_ * T_) / 8), blk, 0, stream>>>(S);

    dim3 gO(E_ / 64, T_ / 256, B_);
    attn_out_kernel<<<gO, blk, 0, stream>>>(S, Vt, out);
}
